// LateralEI_76519137346066
// MI455X (gfx1250) — matrix operand order repaired and checked by reading, not compiled, not run on hardware
//
#include <hip/hip_runtime.h>

// ---------------------------------------------------------------------------
// LateralEI fused difference-of-Gaussians lateral interaction kernel (gfx1250)
//   out = GAIN * rownorm( K_E e^{-d2/2se^2} - K_I e^{-d2/2si^2} ) @ h
// K tiles are generated on the fly in WMMA A-fragment layout; h tiles are
// streamed into double-buffered LDS with global_load_async_to_lds_b128
// (ASYNCcnt) one chunk ahead, then fed to v_wmma_f32_16x16x32_f16.
// B-fragment ds loads + s_wait_dscnt are hand-issued via inline asm with a
// 3-slot rotation so every WMMA runs behind a PARTIAL wait (dscnt<=4), not a
// full drain.
// ---------------------------------------------------------------------------

typedef _Float16 v8h  __attribute__((ext_vector_type(8)));
typedef _Float16 v16h __attribute__((ext_vector_type(16)));
typedef float    v8f  __attribute__((ext_vector_type(8)));

constexpr int   N = 8192;   // rows / j-dimension
constexpr int   D = 512;    // feature dim of h
constexpr int   CHUNKS = N / 32;
constexpr float SIGMA_E = 0.6f;
constexpr float SIGMA_I = 1.2f;
constexpr float K_E  = 0.8f;
constexpr float K_I  = 1.0f;
constexpr float GAIN = 0.05f;
constexpr float LOG2E = 1.44269504088896340736f;
// exp(-d2/(2 s^2)) == exp2(d2 * C)   with C = -log2(e)/(2 s^2)
constexpr float CE2 = -LOG2E / (2.0f * SIGMA_E * SIGMA_E);
constexpr float CI2 = -LOG2E / (2.0f * SIGMA_I * SIGMA_I);

// LDS layout for the main kernel (dynamic shared memory):
//   hbuf: double-buffered h chunk, [col 0..511] x 64B (32 j * f16), padded to
//         80B/col so 16-lane B-fragment ds_load_b128 is bank-conflict free.
constexpr int HBUF_STRIDE = 80;                      // bytes per column row
constexpr int HBUF_BYTES  = D * HBUF_STRIDE;         // 40960 per buffer
constexpr int ZBUF_OFF    = 2 * HBUF_BYTES;          // float4[2][32] = 1KB
constexpr int KBUF_OFF    = ZBUF_OFF + 2 * 32 * 16;  // v8h[2][2][2][32] = 4KB
constexpr int SMEM_BYTES  = KBUF_OFF + 2 * 2 * 2 * 32 * 16;  // 87040

typedef __attribute__((address_space(3))) char lds_char;
__device__ __forceinline__ uint32_t lds_off(const void* p) {
  return (uint32_t)(uintptr_t)(const lds_char*)p;
}

__device__ __forceinline__ void async_wait_all() {
#if __has_builtin(__builtin_amdgcn_s_wait_asynccnt)
  __builtin_amdgcn_s_wait_asynccnt(0);
#else
  asm volatile("s_wait_asynccnt 0x0" ::: "memory");
#endif
}

// ---------------------------------------------------------------------------
// Kernel 1: pack z into SoA float4 {x, y, z, |z|^2}  (one b128 per access)
// ---------------------------------------------------------------------------
__global__ void lei_pack_z(const float* __restrict__ z, float4* __restrict__ z4) {
  int i = blockIdx.x * blockDim.x + threadIdx.x;
  if (i < N) {
    float x = z[3 * i + 0];
    float y = z[3 * i + 1];
    float w = z[3 * i + 2];
    z4[i] = make_float4(x, y, w, x * x + y * y + w * w);
  }
}

// ---------------------------------------------------------------------------
// Kernel 2: per-row scale s_i = GAIN / (sum_j K_ij + 1e-6).
// One wave32 per row; lanes stride j; shuffle reduction.
// ---------------------------------------------------------------------------
__global__ void lei_rowscale(const float4* __restrict__ z4, float* __restrict__ s) {
  const int lane = threadIdx.x & 31;
  const int wave = threadIdx.x >> 5;
  const int row  = blockIdx.x * 8 + wave;
  const float4 zi = z4[row];
  float sum = 0.0f;
  for (int j = lane; j < N; j += 32) {
    float4 zj = z4[j];
    float dot = zi.x * zj.x + zi.y * zj.y + zi.z * zj.z;
    float d2  = fmaxf(__builtin_fmaf(-2.0f, dot, zi.w + zj.w), 0.0f);
    sum += K_E * __builtin_exp2f(d2 * CE2) - K_I * __builtin_exp2f(d2 * CI2);
  }
#pragma unroll
  for (int off = 16; off; off >>= 1) sum += __shfl_xor(sum, off, 32);
  if (lane == 0) s[row] = GAIN / (sum + 1e-6f);
}

// ---------------------------------------------------------------------------
// Kernel 3: h (f32, row-major [j][d]) -> hT (f16, row-major [d][j]).
// LDS 32x32 tile transpose so both sides are coalesced.
// ---------------------------------------------------------------------------
__global__ void lei_transpose_h(const float* __restrict__ h, _Float16* __restrict__ hT) {
  __shared__ _Float16 tile[32][33];
  const int dt = blockIdx.x * 32;   // d tile base
  const int jt = blockIdx.y * 32;   // j tile base
  const int tx = threadIdx.x;       // 0..31
  const int ty = threadIdx.y;       // 0..7
#pragma unroll
  for (int k = 0; k < 4; ++k)
    tile[ty + 8 * k][tx] = (_Float16)h[(size_t)(jt + ty + 8 * k) * D + dt + tx];
  __syncthreads();
#pragma unroll
  for (int k = 0; k < 4; ++k)
    hT[(size_t)(dt + ty + 8 * k) * N + jt + tx] = tile[tx][ty + 8 * k];
}

// ---------------------------------------------------------------------------
// Kernel 4: fused DoG-kernel GEMM.
// Block = 128 threads = 4 wave32: 2 row-groups x 2 col-groups.
// Block output slab: 32 rows x 512 cols. Wave slab: 16 rows x 256 cols
// (16 v8f accumulators). Per 32-wide j chunk:
//   * async-copy next chunk's hT slab (32KB) into LDS (prefetch distance = 1)
//   * each col-group wave produces HALF of its row-group's 16x32 f16 K
//     A-fragment (lane L holds row M = L&15; elems e: K = 8*cg + 16*(L>=16)+e)
//   * halves exchanged through double-buffered LDS; ONE barrier per chunk
//   * 16x v_wmma_f32_16x16x32_f16 with hand-scheduled asm ds loads/waits.
// DScnt invariant: compiler drains DScnt to 0 before each s_barrier (it must,
// for its own kbuf/zbuf ds stores), and our manual waits drain our own loads
// to 0 by the last tile — so manual and compiler waitcnt bookkeeping never
// see each other's operations in flight.
// ---------------------------------------------------------------------------
__global__ void __launch_bounds__(128)
lei_dog_gemm(const float4* __restrict__ z4, const float* __restrict__ s,
             const _Float16* __restrict__ hT, float* __restrict__ out) {
  extern __shared__ char smem[];
  float4* zbuf = (float4*)(smem + ZBUF_OFF);   // [buf][32]
  v8h*    kbuf = (v8h*)(smem + KBUF_OFF);      // [buf][rg][half][lane]

  const int tid  = threadIdx.x;
  const int lane = tid & 31;
  const int wave = tid >> 5;
  const int rg   = wave >> 1;             // row group 0..1  (16 rows each)
  const int cg   = wave & 1;              // col group 0..1  (256 cols each)
  const int i0   = blockIdx.x * 32;       // block row base

  // Per-lane row constants (A-fragment: lane L -> matrix row M = L & 15).
  const int    irow = i0 + rg * 16 + (lane & 15);
  const float4 zi   = z4[irow];
  const float  si   = s[irow];
  const float  kep  = K_E * si;           // fold GAIN/rowsum into DoG weights
  const float  kip  = K_I * si;

  // ---- async h-chunk staging: thread tid copies 16x b128 per chunk ----
  // segment s2: column = s2*32 + (tid>>2), bytes (tid&3)*16 .. +16 of the
  // 64-byte (32 x f16) column row.
  const uint32_t smbase   = lds_off(smem);
  const char*    gcb      = (const char*)hT +
                            (size_t)(tid >> 2) * (N * 2) + (size_t)(tid & 3) * 16;
  const uint32_t lcb      = smbase + (uint32_t)((tid >> 2) * HBUF_STRIDE +
                                                (tid & 3) * 16);
  auto issue_chunk_copies = [&](int chunk) {
    const char* g = gcb + (size_t)chunk * 64;
    uint32_t    l = lcb + (uint32_t)(chunk & 1) * HBUF_BYTES;
#pragma unroll
    for (int s2 = 0; s2 < 16; ++s2) {
      asm volatile("global_load_async_to_lds_b128 %0, %1, off"
                   :: "v"(l), "v"(g) : "memory");
      g += (size_t)32 * (N * 2);   // 32 columns ahead
      l += 32 * HBUF_STRIDE;
    }
  };

  // B-fragment base inside a chunk buffer: lane L -> col n = cg*256 + (L&15),
  // j-halves split at 16 (32B into the 64B column row).
  const int boff = (cg * 256 + (lane & 15)) * HBUF_STRIDE + ((lane & 16) ? 32 : 0);

  const v8f vzero = {0.f, 0.f, 0.f, 0.f, 0.f, 0.f, 0.f, 0.f};
  v8f acc[16];
#pragma unroll
  for (int t = 0; t < 16; ++t) acc[t] = vzero;

  // Prologue: stage z chunk 0, kick off async copy of h chunk 0.
  if (tid < 32) zbuf[lane] = z4[lane];
  issue_chunk_copies(0);
  __syncthreads();

  // This wave's 8 j-offsets inside the chunk start here (consecutive).
  const int zoff = cg * 8 + ((lane & 16) ? 16 : 0);

  for (int c = 0; c < CHUNKS; ++c) {
    const int cur = c & 1;
    const int nxt = cur ^ 1;

    // ---- produce half of the K A-fragment for rows [i0+rg*16, +16) ----
    v8h kh;
#pragma unroll
    for (int e = 0; e < 8; ++e) {
      float4 zj  = zbuf[cur * 32 + zoff + e];                 // ds_load_b128
      float  dot = zi.x * zj.x + zi.y * zj.y + zi.z * zj.z;
      float  d2  = fmaxf(__builtin_fmaf(-2.0f, dot, zi.w + zj.w), 0.0f);
      float  kv  = kep * __builtin_exp2f(d2 * CE2) -
                   kip * __builtin_exp2f(d2 * CI2);
      kh[e] = (_Float16)kv;
    }
    kbuf[((cur * 2 + rg) * 2 + cg) * 32 + lane] = kh;         // ds_store_b128

    if (tid < 32 && c + 1 < CHUNKS) zbuf[nxt * 32 + lane] = z4[(c + 1) * 32 + lane];

    async_wait_all();        // this wave's copies for chunk c have landed
    __syncthreads();         // all waves' copies + kbuf/zbuf visible
                             // (compiler drained DScnt to 0 before barrier)

    // ---- manual DS pipeline: A-fragment + tiles 0,1 (6 ops in flight) ----
    const uint32_t kaddr = smbase +
        (uint32_t)(KBUF_OFF + ((cur * 2 + rg) * 2) * 512 + lane * 16);
    v8h af0, af1;
    asm volatile("ds_load_b128 %0, %2\n\t"
                 "ds_load_b128 %1, %2 offset:512"
                 : "=v"(af0), "=v"(af1)
                 : "v"(kaddr));

    const uint32_t hpo = smbase + (uint32_t)(cur * HBUF_BYTES + boff);
    v8h bl[3][2];
#pragma unroll
    for (int p = 0; p < 2; ++p)
      asm volatile("ds_load_b128 %0, %2 offset:%3\n\t"
                   "ds_load_b128 %1, %2 offset:%4"
                   : "=v"(bl[p][0]), "=v"(bl[p][1])
                   : "v"(hpo), "i"(p * 16 * HBUF_STRIDE),
                     "i"(p * 16 * HBUF_STRIDE + 16));

    // Safe to overwrite hbuf[nxt] now (all waves are past chunk c-1 reads);
    // address-setup VALU executes under the DS-load latency.
    if (c + 1 < CHUNKS) issue_chunk_copies(c + 1);

    // ---- 16 tiles: load(t+2) / wait(dscnt<=4) / wmma(t) ----
    union { v16h v; v8h h[2]; } af;
#pragma unroll
    for (int t = 0; t < 16; ++t) {
      if (t + 2 < 16)
        asm volatile("ds_load_b128 %0, %2 offset:%3\n\t"
                     "ds_load_b128 %1, %2 offset:%4"
                     : "=v"(bl[(t + 2) % 3][0]), "=v"(bl[(t + 2) % 3][1])
                     : "v"(hpo), "i"((t + 2) * 16 * HBUF_STRIDE),
                       "i"((t + 2) * 16 * HBUF_STRIDE + 16));
      // Partial wait; "+v" ties create data deps so neither the WMMA nor any
      // copies of asm-load outputs can be scheduled above the wait.
      if (t == 0) {
        asm volatile("s_wait_dscnt 0x4"
                     : "+v"(bl[0][0]), "+v"(bl[0][1]), "+v"(af0), "+v"(af1)
                     :: "memory");
        af.h[0] = af0;   // K 16*(L>=16) + 0..7
        af.h[1] = af1;   // K 16*(L>=16) + 8..15
      } else if (t < 14) {
        asm volatile("s_wait_dscnt 0x4"
                     : "+v"(bl[t % 3][0]), "+v"(bl[t % 3][1]) :: "memory");
      } else if (t == 14) {
        asm volatile("s_wait_dscnt 0x2"
                     : "+v"(bl[t % 3][0]), "+v"(bl[t % 3][1]) :: "memory");
      } else {
        asm volatile("s_wait_dscnt 0x0"
                     : "+v"(bl[t % 3][0]), "+v"(bl[t % 3][1]) :: "memory");
      }
      union { v16h v; v8h h[2]; } bf;
      bf.h[0] = bl[t % 3][0];
      bf.h[1] = bl[t % 3][1];
      acc[t] = __builtin_amdgcn_wmma_f32_16x16x32_f16(
          false, af.v, false, bf.v, (short)0, acc[t], false, false);
    }
  }

  // ---- epilogue: C/D layout -> out[i][d] (f32) ----
  // VGPR r of a tile holds row M = r (lanes 0-15) or r+8 (lanes 16-31),
  // column N = lane & 15.
  const int orow = i0 + rg * 16 + ((lane & 16) ? 8 : 0);
  const int ocol = cg * 256 + (lane & 15);
#pragma unroll
  for (int t = 0; t < 16; ++t) {
#pragma unroll
    for (int r = 0; r < 8; ++r) {
      out[(size_t)(orow + r) * D + ocol + t * 16] = acc[t][r];
    }
  }
}

// ---------------------------------------------------------------------------
// Launcher. Workspace layout (bytes):
//   [0,       131072) : z4   float4[8192]
//   [131072,  163840) : s    float [8192]
//   [163840, 8552448) : hT   f16   [512][8192]
// ---------------------------------------------------------------------------
extern "C" void kernel_launch(void* const* d_in, const int* in_sizes, int n_in,
                              void* d_out, int out_size, void* d_ws, size_t ws_size,
                              hipStream_t stream) {
  (void)in_sizes; (void)n_in; (void)out_size; (void)ws_size;
  const float* z = (const float*)d_in[0];   // [8192, 3]   f32
  const float* h = (const float*)d_in[1];   // [8192, 512] f32
  float* out = (float*)d_out;               // [8192, 512] f32

  char* ws = (char*)d_ws;
  float4*    z4 = (float4*)(ws);
  float*     sc = (float*)(ws + 131072);
  _Float16*  hT = (_Float16*)(ws + 163840);

  lei_pack_z     <<<(N + 255) / 256, 256, 0, stream>>>(z, z4);
  lei_rowscale   <<<N / 8, 256, 0, stream>>>(z4, sc);
  lei_transpose_h<<<dim3(D / 32, N / 32), dim3(32, 8), 0, stream>>>(h, hT);
  lei_dog_gemm   <<<N / 32, 128, SMEM_BYTES, stream>>>(z4, sc, hT, out);
}
